// dRQA_46995532153133
// MI455X (gfx1250) — matrix operand order repaired and checked by reading, not compiled, not run on hardware
//
#include <hip/hip_runtime.h>

typedef __attribute__((ext_vector_type(16))) _Float16 v16h;
typedef __attribute__((ext_vector_type(8)))  _Float16 v8h;
typedef __attribute__((ext_vector_type(8)))  float    v8f;
typedef __attribute__((ext_vector_type(4)))  int      v4i;

#define N_ROWS 8192
#define DIMK   256
#define TILE   128
#define KCH    32
#define KROW   (KCH + 8)   // 40 f16 = 80B rows: 16B aligned, bank-skewed

// ---- CDNA5 async memory->LDS path (hedged: falls back to load+ds_store) ----
#if defined(__HIP_DEVICE_COMPILE__) && __has_builtin(__builtin_amdgcn_global_load_async_to_lds_b128)
#define USE_ASYNC 1
typedef __attribute__((address_space(1))) v4i gas_v4i;
typedef __attribute__((address_space(3))) v4i las_v4i;
#else
#define USE_ASYNC 0
#endif

__device__ __forceinline__ void wait_async_all() {
#if USE_ASYNC
#if __has_builtin(__builtin_amdgcn_s_wait_asynccnt)
    __builtin_amdgcn_s_wait_asynccnt(0);
#else
    asm volatile("s_wait_asynccnt 0" ::: "memory");
#endif
#endif
}

__device__ __forceinline__ void copy16_to_lds(const _Float16* g, _Float16* l) {
#if USE_ASYNC
    __builtin_amdgcn_global_load_async_to_lds_b128((gas_v4i*)g, (las_v4i*)l, 0, 0);
#else
    *(uint4*)l = *(const uint4*)g;
#endif
}

// ---------------------------------------------------------------- zero output
__global__ void zero_out_kernel(float* __restrict__ out, int n) {
    int i = blockIdx.x * blockDim.x + threadIdx.x;
    if (i < n) out[i] = 0.0f;
}

// ---------------------------------------------- row-normalize f32 -> f16 rows
__global__ __launch_bounds__(256)
void normalize_rows_f16(const float* __restrict__ src, _Float16* __restrict__ dst) {
    const int tid  = threadIdx.x;
    const int lane = tid & 31;
    const int row  = blockIdx.x * 8 + (tid >> 5);

    const float* p = src + (size_t)row * DIMK + lane * 8;
    const float4 u0 = ((const float4*)p)[0];
    const float4 u1 = ((const float4*)p)[1];
    float ss = u0.x*u0.x + u0.y*u0.y + u0.z*u0.z + u0.w*u0.w
             + u1.x*u1.x + u1.y*u1.y + u1.z*u1.z + u1.w*u1.w;
    #pragma unroll
    for (int off = 16; off >= 1; off >>= 1)
        ss += __shfl_xor(ss, off, 32);

    const float scale = 1.0f / fmaxf(sqrtf(ss), 1e-8f);

    v8h h;
    h[0] = (_Float16)(u0.x * scale); h[1] = (_Float16)(u0.y * scale);
    h[2] = (_Float16)(u0.z * scale); h[3] = (_Float16)(u0.w * scale);
    h[4] = (_Float16)(u1.x * scale); h[5] = (_Float16)(u1.y * scale);
    h[6] = (_Float16)(u1.z * scale); h[7] = (_Float16)(u1.w * scale);
    *(v8h*)(dst + (size_t)row * DIMK + lane * 8) = h;
}

// ------------------------------------------------- main GEMM + gaussian + sum
// Block: 128x128 C-tile, 8 waves; each wave owns a 32x64 sub-tile
// (2 A-frags x 4 B-frags -> 8 WMMAs/kstep, 12 ds_load_b128/kstep).
__global__ __launch_bounds__(256)
void cosgauss_wmma(const _Float16* __restrict__ Xh,
                   const _Float16* __restrict__ Yh,
                   float* __restrict__ out) {
    __shared__ _Float16 As[2][TILE][KROW];
    __shared__ _Float16 Bs[2][TILE][KROW];
    __shared__ float    colsum[TILE];

    const int tid  = threadIdx.x;
    const int lane = tid & 31;
    const int wave = tid >> 5;
    const int n0   = blockIdx.x * TILE;
    const int m0   = blockIdx.y * TILE;

    if (tid < TILE) colsum[tid] = 0.0f;

    v8f acc[8] = {};   // acc[t]: rows rbase..+16, acc[t+4]: rows rbase+16..+32

    // staging: per pass each thread copies 16B; 64 rows x 4 chunks of 16B
    const int ldRow = tid >> 2;          // 0..63
    const int ldCol = (tid & 3) * 8;     // f16 offset within 32-wide K chunk

    // WMMA fragment addressing (ISA 7.12.2, wave32 f16 layouts)
    const int col   = lane & 15;
    const int rbase = (wave >> 1) * 32;          // wave's 32-row strip
    const int cbase = (wave & 1) * 64;           // wave's 64-col strip
    const int koffA = (lane < 16) ? 0 : 8;       // A: V0-3 K0-7|8-15, V4-7 K16-23|24-31
    const int koffB = (lane < 16) ? 0 : 8;       // B: V0-3 K0-7|8-15, V4-7 K16-23|24-31

    auto stage = [&](int k0, int buf) {
        #pragma unroll
        for (int r = 0; r < TILE; r += 64) {
            copy16_to_lds(Xh + (size_t)(n0 + ldRow + r) * DIMK + k0 + ldCol,
                          &As[buf][ldRow + r][ldCol]);
            copy16_to_lds(Yh + (size_t)(m0 + ldRow + r) * DIMK + k0 + ldCol,
                          &Bs[buf][ldRow + r][ldCol]);
        }
    };

    auto loadA = [&](int buf, int rowofs) -> v16h {
        const v8h a0 = *(const v8h*)&As[buf][rbase + rowofs + col][koffA];
        const v8h a1 = *(const v8h*)&As[buf][rbase + rowofs + col][koffA + 16];
        v16h a;
        #pragma unroll
        for (int i = 0; i < 8; ++i) { a[i] = a0[i]; a[i + 8] = a1[i]; }
        return a;
    };

    auto loadB = [&](int buf, int t) -> v16h {
        const v8h b0 = *(const v8h*)&Bs[buf][cbase + t * 16 + col][koffB];
        const v8h b1 = *(const v8h*)&Bs[buf][cbase + t * 16 + col][koffB + 16];
        v16h b;
        #pragma unroll
        for (int i = 0; i < 8; ++i) { b[i] = b0[i]; b[i + 8] = b1[i]; }
        return b;
    };

    stage(0, 0);                             // prologue prefetch of chunk 0

    #pragma unroll 2
    for (int c = 0; c < DIMK / KCH; ++c) {
        const int buf = c & 1;
        wait_async_all();                    // this wave's chunk-c copies done
        __syncthreads();                     // all copies visible, prev compute done
        if (c < DIMK / KCH - 1)
            stage((c + 1) * KCH, buf ^ 1);   // overlap next chunk with compute

        const v16h a0 = loadA(buf, 0);
        const v16h a1 = loadA(buf, 16);

        // each B fragment feeds two WMMAs (rows rbase and rbase+16)
        v16h bc = loadB(buf, 0);
        #pragma unroll
        for (int t = 0; t < 4; ++t) {
            v16h bn = bc;
            if (t < 3) bn = loadB(buf, t + 1);
            acc[t] = __builtin_amdgcn_wmma_f32_16x16x32_f16(
                false, a0, false, bc, (short)0, acc[t], false, false);
            acc[t + 4] = __builtin_amdgcn_wmma_f32_16x16x32_f16(
                false, a1, false, bc, (short)0, acc[t + 4], false, false);
            bc = bn;
        }
    }

    // Epilogue: gaussian pdf + column reduction.
    // C/D layout: lane L, vgpr r -> M = r + 8*(L/16), N = L%16.
    #pragma unroll
    for (int t = 0; t < 4; ++t) {
        float s = 0.0f;
        #pragma unroll
        for (int r = 0; r < 8; ++r) {
            const float z0 = (acc[t][r]     - 1.0f) * 20.0f;   // / sigma
            const float z1 = (acc[t + 4][r] - 1.0f) * 20.0f;
            s += __expf(2.0767937f - 0.5f * z0 * z0);          // exp(LOG_NORM - z^2/2)
            s += __expf(2.0767937f - 0.5f * z1 * z1);
        }
        s += __shfl_xor(s, 16, 32);                            // fold M halves (same N)
        if (lane < 16)
            atomicAdd(&colsum[cbase + t * 16 + col], s);       // ds_add_f32 across waves
    }
    __syncthreads();
    if (tid < TILE) atomicAdd(&out[m0 + tid], colsum[tid]);
}

// ----------------------------------------------------------------- host entry
extern "C" void kernel_launch(void* const* d_in, const int* in_sizes, int n_in,
                              void* d_out, int out_size, void* d_ws, size_t ws_size,
                              hipStream_t stream) {
    const float* Ex = (const float*)d_in[0];
    const float* Ey = (const float*)d_in[1];
    float* out = (float*)d_out;

    _Float16* Xh = (_Float16*)d_ws;                 // 4MB
    _Float16* Yh = Xh + (size_t)N_ROWS * DIMK;      // 4MB

    zero_out_kernel<<<(N_ROWS + 255) / 256, 256, 0, stream>>>(out, N_ROWS);
    normalize_rows_f16<<<N_ROWS / 8, 256, 0, stream>>>(Ex, Xh);
    normalize_rows_f16<<<N_ROWS / 8, 256, 0, stream>>>(Ey, Yh);

    dim3 grid(N_ROWS / TILE, N_ROWS / TILE);        // 64 x 64 tiles
    cosgauss_wmma<<<grid, 256, 0, stream>>>(Xh, Yh, out);
}
